// StructureModule_48576080117837
// MI455X (gfx1250) — matrix operand order repaired and checked by reading, not compiled, not run on hardware
//
#include <hip/hip_runtime.h>
#include <hip/hip_bf16.h>
#include <math.h>
#include <stdint.h>

#define Bh   2
#define Nn   384
#define CS   384
#define CZ   128
#define HI   12
#define PQK  4
#define PV   8
#define HTR  8
#define DTR  512
#define NB   (Bh*Nn)          // 768
#define ADA6 (6*CS)           // 2304

typedef __bf16 bf16x16 __attribute__((ext_vector_type(16)));
typedef float  f32x8   __attribute__((ext_vector_type(8)));
typedef int    v4i_t   __attribute__((vector_size(16)));     // builtin arg type
union FragB { bf16x16 v; unsigned int u[8]; };
union PK2   { __bf16 h[2]; unsigned int u; };

__device__ inline unsigned pk2(float x, float y) {
  PK2 p; p.h[0] = (__bf16)x; p.h[1] = (__bf16)y; return p.u;
}

// CDNA5 async global->LDS path (raw bf16 weight tiles), with sync fallback.
#if defined(__HIP_DEVICE_COMPILE__) && \
    __has_builtin(__builtin_amdgcn_global_load_async_to_lds_b128) && \
    __has_builtin(__builtin_amdgcn_s_wait_asynccnt)
#define ASYNC_LDS 1
#endif

// ---------------------------------------------------------------------------
// Generic batched GEMM:  C = alpha * (A @ B) + bias  (optional ReLU)
// A: f32 [M x K] row-stride lda (converted to bf16 while staging).
// TB=float : B f32, transB=0 -> B[k*ldb+n], 1 -> B[n*ldb+k]
// TB=__bf16: B pre-converted & pre-transposed weights, layout [n][k], ldb=K,
//            staged into LDS via async DMA (raw copy) on full tiles.
// batch z decomposed: bo = z/divi, bi = z%divi; base += bo*s?o + bi*s?i
// ---------------------------------------------------------------------------
template <typename TB>
struct GemmP {
  const float* A; const TB* B; const float* bias; float* C;
  long long sAo, sAi, sBo, sBi, sCo, sCi;
  int divi, lda, ldb, ldc, M, N, K, transB, relu;
  float alpha;
};

template <typename TB>
__global__ __launch_bounds__(128) void gemm_wmma(GemmP<TB> p) {
  __shared__ __bf16 As[64][40];   // [m][k], row stride 80B (16B aligned)
  __shared__ __bf16 Bs[64][40];   // [n][k]
  const int bm = blockIdx.x * 64, bn = blockIdx.y * 64;
  const int zb = blockIdx.z;
  const int bo = zb / p.divi, bi = zb % p.divi;
  const float* A = p.A + bo * p.sAo + bi * p.sAi;
  const TB*    B = p.B + bo * p.sBo + bi * p.sBi;
  float*       C = p.C + bo * p.sCo + bi * p.sCi;
  const int tid = threadIdx.x, lane = tid & 31, wave = tid >> 5;
  const int wm = (wave & 1) * 32, wn = (wave >> 1) * 32;
  const int L = lane >> 4, rr = lane & 15;
  f32x8 acc[2][2] = {};

  for (int k0 = 0; k0 < p.K; k0 += 32) {
    const bool fullK = (k0 + 32 <= p.K);

    // ---- stage A tile (f32 -> bf16) ----
    if (fullK && bm + 64 <= p.M) {
      #pragma unroll
      for (int it = 0; it < 2; ++it) {
        int slot = tid + it * 128;             // 256 slots: 64 rows x 4 chunks
        int r = slot >> 2, ko = (slot & 3) * 8;
        const float* src = &A[(long long)(bm + r) * p.lda + k0 + ko];
        float4 a0 = *(const float4*)src;
        float4 a1 = *(const float4*)(src + 4);
        unsigned* d = (unsigned*)&As[r][ko];
        d[0] = pk2(a0.x, a0.y); d[1] = pk2(a0.z, a0.w);
        d[2] = pk2(a1.x, a1.y); d[3] = pk2(a1.z, a1.w);
      }
    } else {
      for (int idx = tid; idx < 64*32; idx += 128) {
        int r = idx >> 5, c = idx & 31;
        int gm = bm + r, gk = k0 + c;
        float v = 0.f;
        if (gm < p.M && gk < p.K) v = A[(long long)gm * p.lda + gk];
        As[r][c] = (__bf16)v;
      }
    }

    // ---- stage B tile ----
    if constexpr (sizeof(TB) == 2) {          // bf16 weights, [n][k], ldb=K
      if (fullK && bn + 64 <= p.N) {
#ifdef ASYNC_LDS
        #pragma unroll
        for (int it = 0; it < 2; ++it) {
          int slot = tid + it * 128;
          int r = slot >> 2, ko = (slot & 3) * 8;
          const TB* src = &B[(long long)(bn + r) * p.ldb + k0 + ko];
          __builtin_amdgcn_global_load_async_to_lds_b128(
              (v4i_t*)src, (v4i_t*)&Bs[r][ko], 0, 0);
        }
        __builtin_amdgcn_s_wait_asynccnt(0);
#else
        #pragma unroll
        for (int it = 0; it < 2; ++it) {
          int slot = tid + it * 128;
          int r = slot >> 2, ko = (slot & 3) * 8;
          *(uint4*)&Bs[r][ko] = *(const uint4*)&B[(long long)(bn + r) * p.ldb + k0 + ko];
        }
#endif
      } else {
        for (int idx = tid; idx < 64*32; idx += 128) {
          int n = idx >> 5, c = idx & 31;
          int gn = bn + n, gk = k0 + c;
          TB v = (TB)0.f;
          if (gn < p.N && gk < p.K) v = B[(long long)gn * p.ldb + gk];
          Bs[n][c] = v;
        }
      }
    } else {                                   // f32 B
      if (fullK && bn + 64 <= p.N) {
        if (p.transB) {
          #pragma unroll
          for (int it = 0; it < 2; ++it) {
            int slot = tid + it * 128;
            int r = slot >> 2, ko = (slot & 3) * 8;
            const float* src = &B[(long long)(bn + r) * p.ldb + k0 + ko];
            float4 a0 = *(const float4*)src;
            float4 a1 = *(const float4*)(src + 4);
            unsigned* d = (unsigned*)&Bs[r][ko];
            d[0] = pk2(a0.x, a0.y); d[1] = pk2(a0.z, a0.w);
            d[2] = pk2(a1.x, a1.y); d[3] = pk2(a1.z, a1.w);
          }
        } else {                               // k-major: coalesce along n
          for (int idx = tid; idx < 64*32; idx += 128) {
            int n = idx & 63, k = idx >> 6;
            Bs[n][k] = (__bf16)B[(long long)(k0 + k) * p.ldb + (bn + n)];
          }
        }
      } else {
        for (int idx = tid; idx < 64*32; idx += 128) {
          int n, k;
          if (p.transB) { n = idx >> 5; k = idx & 31; }
          else          { n = idx & 63; k = idx >> 6; }
          int gn = bn + n, gk = k0 + k;
          float v = 0.f;
          if (gn < p.N && gk < p.K)
            v = p.transB ? B[(long long)gn * p.ldb + gk]
                         : B[(long long)gk * p.ldb + gn];
          Bs[n][k] = (__bf16)v;
        }
      }
    }

    if (k0 + 32 < p.K) {                       // speculative prefetch, next A tile
      int gm = bm + (tid & 63);
      if (gm < p.M) __builtin_prefetch(&A[(long long)gm * p.lda + k0 + 32], 0, 0);
    }
    __syncthreads();

    FragB fa[2], fb[2];
    #pragma unroll
    for (int i = 0; i < 2; ++i) {              // A frag: 16x32 bf16 layout
      int row = wm + i*16 + rr;
      #pragma unroll
      for (int v8 = 0; v8 < 8; ++v8) {
        int k = 2*v8 + ((v8 >= 4) ? 8 : 0) + 8*L;   // ISA 7.12.2
        fa[i].u[v8] = *(const unsigned int*)&As[row][k];
      }
    }
    #pragma unroll
    for (int j = 0; j < 2; ++j) {              // B frag: 32x16 bf16 layout
      int col = wn + j*16 + rr;
      #pragma unroll
      for (int v8 = 0; v8 < 8; ++v8) {
        int k = 2*v8 + ((v8 >= 4) ? 8 : 0) + 8*L;
        fb[j].u[v8] = *(const unsigned int*)&Bs[col][k];
      }
    }
    #pragma unroll
    for (int i = 0; i < 2; ++i)
      #pragma unroll
      for (int j = 0; j < 2; ++j)
        acc[i][j] = __builtin_amdgcn_wmma_f32_16x16x32_bf16(
            false, fa[i].v, false, fb[j].v, (short)0, acc[i][j], false, false);
    __syncthreads();
  }

  #pragma unroll
  for (int i = 0; i < 2; ++i) {
    #pragma unroll
    for (int j = 0; j < 2; ++j) {
      int col = bn + wn + j*16 + rr;
      if (col >= p.N) continue;
      float bv = p.bias ? p.bias[col] : 0.f;
      #pragma unroll
      for (int r8 = 0; r8 < 8; ++r8) {
        int row = bm + wm + i*16 + 8*L + r8;   // C layout: M = vgpr + 8*L
        if (row < p.M) {
          float v = acc[i][j][r8] * p.alpha + bv;
          if (p.relu) v = fmaxf(v, 0.f);
          C[(long long)row * p.ldc + col] = v;
        }
      }
    }
  }
}

// ---------------------------------------------------------------------------
// Reduction helpers (blockDim == 128, 4 waves of 32)
// ---------------------------------------------------------------------------
__device__ inline void blk_sum2(float& a, float& b, float* sa, float* sb) {
  for (int o = 16; o; o >>= 1) { a += __shfl_xor(a, o); b += __shfl_xor(b, o); }
  __syncthreads();
  int w = threadIdx.x >> 5;
  if ((threadIdx.x & 31) == 0) { sa[w] = a; sb[w] = b; }
  __syncthreads();
  a = sa[0] + sa[1] + sa[2] + sa[3];
  b = sb[0] + sb[1] + sb[2] + sb[3];
}
__device__ inline float blk_max(float v, float* sm) {
  for (int o = 16; o; o >>= 1) v = fmaxf(v, __shfl_xor(v, o));
  __syncthreads();
  int w = threadIdx.x >> 5;
  if ((threadIdx.x & 31) == 0) sm[w] = v;
  __syncthreads();
  return fmaxf(fmaxf(sm[0], sm[1]), fmaxf(sm[2], sm[3]));
}
__device__ inline float blk_sum1(float v, float* sm) {
  for (int o = 16; o; o >>= 1) v += __shfl_xor(v, o);
  __syncthreads();
  int w = threadIdx.x >> 5;
  if ((threadIdx.x & 31) == 0) sm[w] = v;
  __syncthreads();
  return sm[0] + sm[1] + sm[2] + sm[3];
}

// ---------------------------------------------------------------------------
// Small kernels
// ---------------------------------------------------------------------------
__global__ void k_cvt_wt(const float* w, __bf16* o, int K, int N) {
  long long idx = (long long)blockIdx.x * 256 + threadIdx.x;   // out-coalesced
  if (idx >= (long long)K * N) return;
  int k = (int)(idx % K); long long n = idx / K;
  o[n * K + k] = (__bf16)w[(long long)k * N + n];              // transpose -> [n][k]
}

__global__ void k_silu(const float* t, float* o, int n) {
  int i = blockIdx.x * 256 + threadIdx.x;
  if (i < n) { float x = t[i]; o[i] = x / (1.f + __expf(-x)); }
}

__global__ void k_ada(const float* st, const float* w, const float* b, float* ada) {
  int idx = blockIdx.x * 256 + threadIdx.x;
  if (idx >= Bh * ADA6) return;
  int j = idx % ADA6, bb = idx / ADA6;
  float s = b[j];
  for (int k = 0; k < CS; ++k) s += st[bb*CS + k] * w[k*ADA6 + j];
  ada[idx] = s;
}

// LN (no affine) + adaLN modulation: out = ln(x)*(1+sc) + sh
__global__ __launch_bounds__(128) void k_ln_mod(const float* x, const float* ada,
                                                int shOff, int scOff, float* out,
                                                float eps) {
  int r = blockIdx.x, b = r / Nn;
  const float* xr = x + (long long)r * CS;
  float xs[3]; float s1 = 0.f, s2 = 0.f;
  for (int t = 0; t < 3; ++t) {
    int c = threadIdx.x + t*128; float v = xr[c];
    xs[t] = v; s1 += v; s2 += v*v;
  }
  __shared__ float sa[4], sb[4];
  blk_sum2(s1, s2, sa, sb);
  float m = s1 / CS, var = s2 / CS - m*m, rs = rsqrtf(var + eps);
  for (int t = 0; t < 3; ++t) {
    int c = threadIdx.x + t*128;
    float y = (xs[t] - m) * rs;
    out[(long long)r*CS + c] = y * (1.f + ada[b*ADA6 + scOff + c]) + ada[b*ADA6 + shOff + c];
  }
}

// s1 = s + g_m*mask*ipa ; out = ln(s1)*g+b ; also copied into s_cat[:,0:384]
__global__ __launch_bounds__(128) void k_gate_add_ln(const float* s, const float* ipa,
    const float* ada, const float* mask, const float* g, const float* be,
    float* spost, float* scat) {
  int r = blockIdx.x, b = r / Nn;
  float mk = mask[r];
  float xs[3]; float s1 = 0.f, s2 = 0.f;
  for (int t = 0; t < 3; ++t) {
    int c = threadIdx.x + t*128;
    float v = s[(long long)r*CS + c] + ada[b*ADA6 + 2*CS + c] * mk * ipa[(long long)r*CS + c];
    xs[t] = v; s1 += v; s2 += v*v;
  }
  __shared__ float sa[4], sb[4];
  blk_sum2(s1, s2, sa, sb);
  float m = s1 / CS, var = s2 / CS - m*m, rs = rsqrtf(var + 1e-6f);
  for (int t = 0; t < 3; ++t) {
    int c = threadIdx.x + t*128;
    float y = (xs[t] - m) * rs * g[c] + be[c];
    spost[(long long)r*CS + c] = y;
    scat[(long long)r*DTR + c] = y;
  }
}

// x = ln(x + y) * g + b   (C = 512, eps 1e-5, in place)
__global__ __launch_bounds__(128) void k_add_ln(float* x, const float* y,
                                                const float* g, const float* be) {
  int r = blockIdx.x;
  float xs[4]; float s1 = 0.f, s2 = 0.f;
  for (int t = 0; t < 4; ++t) {
    int c = threadIdx.x + t*128;
    float v = x[(long long)r*DTR + c] + y[(long long)r*DTR + c];
    xs[t] = v; s1 += v; s2 += v*v;
  }
  __shared__ float sa[4], sb[4];
  blk_sum2(s1, s2, sa, sb);
  float m = s1 / DTR, var = s2 / DTR - m*m, rs = rsqrtf(var + 1e-5f);
  for (int t = 0; t < 4; ++t) {
    int c = threadIdx.x + t*128;
    x[(long long)r*DTR + c] = (xs[t] - m) * rs * g[c] + be[c];
  }
}

// d_out = ln(x)*g+b * mask   (C=384, eps 1e-5)
__global__ __launch_bounds__(128) void k_final_ln(const float* x, const float* g,
    const float* be, const float* mask, float* out) {
  int r = blockIdx.x;
  float mk = mask[r];
  float xs[3]; float s1 = 0.f, s2 = 0.f;
  for (int t = 0; t < 3; ++t) {
    int c = threadIdx.x + t*128; float v = x[(long long)r*CS + c];
    xs[t] = v; s1 += v; s2 += v*v;
  }
  __shared__ float sa[4], sb[4];
  blk_sum2(s1, s2, sa, sb);
  float m = s1 / CS, var = s2 / CS - m*m, rs = rsqrtf(var + 1e-5f);
  for (int t = 0; t < 3; ++t) {
    int c = threadIdx.x + t*128;
    out[(long long)r*CS + c] = ((xs[t] - m) * rs * g[c] + be[c]) * mk;
  }
}

// points -> global frame. raw: [row][x*P + p], out: [row][p*3 + i]
__global__ void k_to_global(const float* raw, const float* rot, const float* trans,
                            float* out, int P) {
  int idx = blockIdx.x * 256 + threadIdx.x;
  if (idx >= NB * P) return;
  int p = idx % P, row = idx / P;
  const float* R = rot + (long long)row * 9;
  const float* tr = trans + (long long)row * 3;
  long long base = (long long)row * 3 * P;
  float x = raw[base + p], y = raw[base + P + p], z = raw[base + 2*P + p];
  long long ob = base + (long long)p * 3;
  out[ob + 0] = R[0]*x + R[1]*y + R[2]*z + tr[0];
  out[ob + 1] = R[3]*x + R[4]*y + R[5]*z + tr[1];
  out[ob + 2] = R[6]*x + R[7]*y + R[8]*z + tr[2];
}

// Build augmented attention features (fold q.k and point-cross term into one K=32 WMMA)
__global__ void k_feat(const float* q, const float* kv, const float* qpts,
                       const float* kvpts, const float* head_w,
                       float* af, float* bf, float* kn) {
  int idx = blockIdx.x * 256 + threadIdx.x;           // (b*HI+h)*Nn + i
  if (idx >= Bh * HI * Nn) return;
  int i = idx % Nn, h = (idx / Nn) % HI, b = idx / (Nn * HI);
  long long row = (long long)(b * Nn + i);
  float hw = log1pf(__expf(head_w[h]));               // softplus
  const float S1 = 0.14433756729740643f;              // sqrt(1/(3*16))
  const float C2 = 0.13608276348795434f;              // sqrt(1/(3*18))
  float* A  = af + (long long)idx * 32;
  float* Bv = bf + (long long)idx * 32;
  for (int c = 0; c < 16; ++c) {
    A[c]  = q[row*192 + h*16 + c] * S1;
    Bv[c] = kv[row*384 + h*32 + c];
  }
  float cA = hw * C2, knv = 0.f;
  for (int e = 0; e < 12; ++e) {
    float qp = qpts[row*144 + h*12 + e];              // Pq points of head h
    float kp = kvpts[row*432 + h*36 + e];             // first 4 points = k_pts
    A[16 + e] = qp * cA;
    Bv[16 + e] = kp;
    knv += kp * kp;
  }
  for (int e = 28; e < 32; ++e) { A[e] = 0.f; Bv[e] = 0.f; }
  kn[idx] = -0.5f * C2 * hw * knv;                    // column bias (row term softmax-invariant)
}

// IPA softmax: scores += kn[j] + sqrt(1/3)*bz + INF*(m_i*m_j - 1), softmax over j
__global__ __launch_bounds__(128) void k_softmax_ipa(float* sc, const float* kn,
    const float* bz, const float* mask) {
  int r = blockIdx.x;                                  // (b*HI+h)*Nn + i
  int i = r % Nn, h = (r / Nn) % HI, b = r / (Nn * HI);
  float* row = sc + (long long)r * Nn;
  float mi = mask[b*Nn + i];
  float v[3]; float mx = -1e30f;
  for (int t = 0; t < 3; ++t) {
    int j = threadIdx.x + t*128;
    float x = row[j] + kn[(b*HI + h)*Nn + j]
            + 0.57735026919f * bz[((long long)(b*Nn + i)*Nn + j)*HI + h]
            + 100000.0f * (mi * mask[b*Nn + j] - 1.0f);
    v[t] = x; mx = fmaxf(mx, x);
  }
  __shared__ float sm[4];
  mx = blk_max(mx, sm);
  float s = 0.f;
  for (int t = 0; t < 3; ++t) { v[t] = __expf(v[t] - mx); s += v[t]; }
  s = blk_sum1(s, sm);
  float inv = 1.f / s;
  for (int t = 0; t < 3; ++t) row[threadIdx.x + t*128] = v[t] * inv;
}

// transformer softmax with mask bias
__global__ __launch_bounds__(128) void k_softmax_mask(float* sc, const float* mask) {
  int r = blockIdx.x;                                  // (b*HTR+h)*Nn + i
  int b = r / (HTR * Nn);
  float* row = sc + (long long)r * Nn;
  float v[3]; float mx = -1e30f;
  for (int t = 0; t < 3; ++t) {
    int j = threadIdx.x + t*128;
    float x = row[j] + (mask[b*Nn + j] == 0.f ? -1e9f : 0.f);
    v[t] = x; mx = fmaxf(mx, x);
  }
  __shared__ float sm[4];
  mx = blk_max(mx, sm);
  float s = 0.f;
  for (int t = 0; t < 3; ++t) { v[t] = __expf(v[t] - mx); s += v[t]; }
  s = blk_sum1(s, sm);
  float inv = 1.f / s;
  for (int t = 0; t < 3; ++t) row[threadIdx.x + t*128] = v[t] * inv;
}

// cat assembly
__global__ void k_cat_o(const float* oh, float* cat) {
  int e = blockIdx.x * 256 + threadIdx.x;
  if (e >= NB * 192) return;
  int row = e / 192, c = e % 192, h = c >> 4, cc = c & 15;
  int b = row / Nn, n = row % Nn;
  cat[(long long)row*2112 + c] = oh[((long long)(b*HI + h)*Nn + n)*16 + cc];
}
__global__ void k_cat_opair(const float* op, float* cat) {
  int e = blockIdx.x * 256 + threadIdx.x;
  if (e >= NB * 1536) return;
  int row = e / 1536, c = e % 1536;
  cat[(long long)row*2112 + 576 + c] = op[(long long)row*1536 + c];
}
__global__ void k_cat_pts(const float* optg, const float* rot, const float* trans,
                          float* cat) {
  int idx = blockIdx.x * 256 + threadIdx.x;            // (row, h, p)
  if (idx >= NB * HI * PV) return;
  int p = idx % PV, h = (idx / PV) % HI, row = idx / (PV * HI);
  int b = row / Nn, n = row % Nn;
  const float* src = optg + ((long long)(b*HI + h)*Nn + n)*24 + p*3;
  const float* R = rot + (long long)row * 9;
  float gx = src[0] - trans[row*3 + 0];
  float gy = src[1] - trans[row*3 + 1];
  float gz = src[2] - trans[row*3 + 2];
  float lx = R[0]*gx + R[3]*gy + R[6]*gz;              // R^T
  float ly = R[1]*gx + R[4]*gy + R[7]*gz;
  float lz = R[2]*gx + R[5]*gy + R[8]*gz;
  float nr = sqrtf(lx*lx + ly*ly + lz*lz + 1e-8f);
  long long base = (long long)row * 2112;
  int hp = h * PV + p;
  cat[base + 192 + hp] = lx; cat[base + 288 + hp] = ly;
  cat[base + 384 + hp] = lz; cat[base + 480 + hp] = nr;
}

__global__ void k_add(const float* a, const float* b, float* c, int n) {
  int i = blockIdx.x * 256 + threadIdx.x;
  if (i < n) c[i] = a[i] + b[i];
}
__global__ void k_nt_final(const float* s2, const float* h, const float* a3,
                           const float* ada, float* s3) {
  int e = blockIdx.x * 256 + threadIdx.x;
  if (e >= NB * CS) return;
  int c = e % CS, b = e / (Nn * CS);
  s3[e] = s2[e] + ada[b*ADA6 + 5*CS + c] * (h[e] + a3[e]);
}

__global__ void k_rigid(const float* upd, const float* rot, const float* trans,
                        float* orot, float* otr) {
  int r = blockIdx.x * 256 + threadIdx.x;
  if (r >= NB) return;
  const float* u = upd + (long long)r * 6;
  float a = 1.f, b = u[0], c = u[1], d = u[2];
  float inv = rsqrtf(a*a + b*b + c*c + d*d);
  a *= inv; b *= inv; c *= inv; d *= inv;
  float rq[9] = { a*a + b*b - c*c - d*d, 2.f*(b*c - a*d),      2.f*(b*d + a*c),
                  2.f*(b*c + a*d),       a*a - b*b + c*c - d*d, 2.f*(c*d - a*b),
                  2.f*(b*d - a*c),       2.f*(c*d + a*b),       a*a - b*b - c*c + d*d };
  const float* R = rot + (long long)r * 9;
  for (int i = 0; i < 3; ++i)
    for (int k = 0; k < 3; ++k)
      orot[(long long)r*9 + i*3 + k] =
          R[i*3+0]*rq[0*3+k] + R[i*3+1]*rq[1*3+k] + R[i*3+2]*rq[2*3+k];
  for (int i = 0; i < 3; ++i)
    otr[(long long)r*3 + i] = R[i*3+0]*u[3] + R[i*3+1]*u[4] + R[i*3+2]*u[5]
                            + trans[(long long)r*3 + i];
}

__global__ void k_copy4(const float4* in, float4* out, long long n) {
  long long i = (long long)blockIdx.x * 256 + threadIdx.x;
  if (i < n) out[i] = in[i];
}

// ---------------------------------------------------------------------------
// Host side
// ---------------------------------------------------------------------------
template <typename TB>
static inline void gemm_t(hipStream_t st,
    const float* A, long long sAo, long long sAi, int lda,
    const TB* B, long long sBo, long long sBi, int ldb, int transB,
    const float* bias, float* C, long long sCo, long long sCi, int ldc,
    int M, int N, int K, int batches, int divi, float alpha, int relu) {
  GemmP<TB> p{A, B, bias, C, sAo, sAi, sBo, sBi, sCo, sCi,
              divi, lda, ldb, ldc, M, N, K, transB, relu, alpha};
  dim3 g((M + 63) / 64, (N + 63) / 64, batches);
  gemm_wmma<TB><<<g, 128, 0, st>>>(p);
}
static inline int gr(long long n) { return (int)((n + 255) / 256); }

extern "C" void kernel_launch(void* const* d_in, const int* in_sizes, int n_in,
                              void* d_out, int out_size, void* d_ws, size_t ws_size,
                              hipStream_t stream) {
  (void)in_sizes; (void)n_in; (void)out_size; (void)ws_size;
  const float* s      = (const float*)d_in[0];
  const float* s_init = (const float*)d_in[1];
  const float* z      = (const float*)d_in[2];
  const float* rot    = (const float*)d_in[3];
  const float* trans  = (const float*)d_in[4];
  const float* t_emb  = (const float*)d_in[5];
  const float* dssp   = (const float*)d_in[6];
  const float* mask   = (const float*)d_in[7];
  const float* ada_w  = (const float*)d_in[8];
  const float* ada_b  = (const float*)d_in[9];
  const float* q_w    = (const float*)d_in[10];
  const float* q_b    = (const float*)d_in[11];
  const float* kv_w   = (const float*)d_in[12];
  const float* kv_b   = (const float*)d_in[13];
  const float* qp_w   = (const float*)d_in[14];
  const float* qp_b   = (const float*)d_in[15];
  const float* kvp_w  = (const float*)d_in[16];
  const float* kvp_b  = (const float*)d_in[17];
  const float* b_w    = (const float*)d_in[18];
  const float* b_b    = (const float*)d_in[19];
  const float* head_w = (const float*)d_in[20];
  const float* out_w  = (const float*)d_in[21];
  const float* out_b  = (const float*)d_in[22];
  const float* ln_ipa_g = (const float*)d_in[23];
  const float* ln_ipa_b = (const float*)d_in[24];
  const float* skip_w = (const float*)d_in[25];
  const float* skip_b = (const float*)d_in[26];
  const float* sd_w   = (const float*)d_in[27];
  const float* sd_b   = (const float*)d_in[28];
  const float* tr_qkv_w = (const float*)d_in[29];
  const float* tr_qkv_b = (const float*)d_in[30];
  const float* tr_o_w   = (const float*)d_in[31];
  const float* tr_o_b   = (const float*)d_in[32];
  const float* tr_ln1_g = (const float*)d_in[33];
  const float* tr_ln1_b = (const float*)d_in[34];
  const float* tr_ff1_w = (const float*)d_in[35];
  const float* tr_ff1_b = (const float*)d_in[36];
  const float* tr_ff2_w = (const float*)d_in[37];
  const float* tr_ff2_b = (const float*)d_in[38];
  const float* tr_ln2_g = (const float*)d_in[39];
  const float* tr_ln2_b = (const float*)d_in[40];
  const float* pt_w   = (const float*)d_in[41];
  const float* pt_b   = (const float*)d_in[42];
  const float* nt1_w  = (const float*)d_in[43];
  const float* nt1_b  = (const float*)d_in[44];
  const float* nt2_w  = (const float*)d_in[45];
  const float* nt2_b  = (const float*)d_in[46];
  const float* nt3_w  = (const float*)d_in[47];
  const float* nt3_b  = (const float*)d_in[48];
  const float* ln_tr_g = (const float*)d_in[49];
  const float* ln_tr_b = (const float*)d_in[50];
  const float* bb_w   = (const float*)d_in[51];
  const float* bb_b   = (const float*)d_in[52];

  float* dout = (float*)d_out;
  const long long OUT_Z = (long long)NB * CS;                       // 294912
  const long long Z_CNT = (long long)Bh * Nn * Nn * CZ;             // 37748736
  const long long OUT_ROT = OUT_Z + Z_CNT;
  const long long OUT_TR  = OUT_ROT + (long long)NB * 9;

  // workspace bump allocator (256B aligned)
  char* wsb = (char*)d_ws; size_t off = 0;
  auto alloc = [&](long long nflt) -> float* {
    float* p = (float*)(wsb + off);
    off += (size_t)(((nflt * 4 + 255) / 256) * 256);
    return p;
  };
  auto allocB = [&](long long nh) -> __bf16* {
    __bf16* p = (__bf16*)(wsb + off);
    off += (size_t)(((nh * 2 + 255) / 256) * 256);
    return p;
  };

  float* stb   = alloc(Bh * CS);
  float* adab  = alloc(Bh * ADA6);
  float* s_mod = alloc((long long)NB * CS);
  float* qb    = alloc((long long)NB * 192);
  float* kvb   = alloc((long long)NB * 384);
  float* qpb   = alloc((long long)NB * 144);
  float* kvpb  = alloc((long long)NB * 432);
  float* qpts  = alloc((long long)NB * 144);
  float* kvpts = alloc((long long)NB * 432);
  float* bzb   = alloc((long long)Bh * Nn * Nn * HI);               // 3.54M
  float* afeat = alloc((long long)Bh * HI * Nn * 32);
  float* bfeat = alloc((long long)Bh * HI * Nn * 32);
  float* knb   = alloc((long long)Bh * HI * Nn);
  float* scores= alloc((long long)Bh * HI * Nn * Nn);               // 3.54M
  float* oh    = alloc((long long)Bh * HI * Nn * 16);
  float* optg  = alloc((long long)Bh * HI * Nn * 24);
  float* opair = alloc((long long)NB * HI * CZ);
  float* cat   = alloc((long long)NB * 2112);
  float* ipaout= alloc((long long)NB * CS);
  float* spost = alloc((long long)NB * CS);
  float* scat  = alloc((long long)NB * DTR);
  float* attno = alloc((long long)NB * DTR);
  float* t1    = alloc((long long)NB * DTR);
  float* t2    = alloc((long long)NB * DTR);
  float* ptout = alloc((long long)NB * CS);
  float* s2b   = alloc((long long)NB * CS);
  float* hb    = alloc((long long)NB * CS);
  float* a1    = alloc((long long)NB * CS);
  float* a2    = alloc((long long)NB * CS);
  float* a3    = alloc((long long)NB * CS);
  float* updb  = alloc((long long)NB * 6);
  // bf16 weight copies, pre-transposed to [n][k]
  __bf16* wq   = allocB(CS * 192);
  __bf16* wkv  = allocB(CS * 384);
  __bf16* wqp  = allocB(CS * 144);
  __bf16* wkvp = allocB(CS * 432);
  __bf16* wbz  = allocB(CZ * HI);
  __bf16* wout = allocB((long long)2112 * CS);
  __bf16* wskip= allocB(CS * 64);
  __bf16* wsd  = allocB(8 * 64);
  __bf16* wqkv = allocB((long long)2 * DTR * 1536);
  __bf16* wo   = allocB((long long)2 * DTR * DTR);
  __bf16* wf1  = allocB((long long)2 * DTR * DTR);
  __bf16* wf2  = allocB((long long)2 * DTR * DTR);
  __bf16* wpt  = allocB(DTR * CS);
  __bf16* wn1  = allocB(CS * CS);
  __bf16* wn2  = allocB(CS * CS);
  __bf16* wn3  = allocB(CS * CS);
  __bf16* wbb  = allocB(CS * 6);
  float* att   = scores;   // alias: transformer attention reuses IPA score region
  float* qkvb  = bzb;      // alias: bz dead after IPA softmax (1.18M < 3.54M)

  // ---- one-time weight convert + transpose (bf16, [n][k]) ----
  auto cvt = [&](const float* w, __bf16* o, int K, int N) {
    k_cvt_wt<<<gr((long long)K * N), 256, 0, stream>>>(w, o, K, N);
  };
  cvt(q_w, wq, CS, 192);      cvt(kv_w, wkv, CS, 384);
  cvt(qp_w, wqp, CS, 144);    cvt(kvp_w, wkvp, CS, 432);
  cvt(b_w, wbz, CZ, HI);      cvt(out_w, wout, 2112, CS);
  cvt(skip_w, wskip, CS, 64); cvt(sd_w, wsd, 8, 64);
  for (int l = 0; l < 2; ++l) {
    cvt(tr_qkv_w + (long long)l*DTR*1536, wqkv + (long long)l*DTR*1536, DTR, 1536);
    cvt(tr_o_w   + (long long)l*DTR*DTR,  wo   + (long long)l*DTR*DTR,  DTR, DTR);
    cvt(tr_ff1_w + (long long)l*DTR*DTR,  wf1  + (long long)l*DTR*DTR,  DTR, DTR);
    cvt(tr_ff2_w + (long long)l*DTR*DTR,  wf2  + (long long)l*DTR*DTR,  DTR, DTR);
  }
  cvt(pt_w, wpt, DTR, CS);
  cvt(nt1_w, wn1, CS, CS); cvt(nt2_w, wn2, CS, CS); cvt(nt3_w, wn3, CS, CS);
  cvt(bb_w, wbb, CS, 6);

  // ---- adaLN params ----
  k_silu<<<gr(Bh*CS), 256, 0, stream>>>(t_emb, stb, Bh*CS);
  k_ada<<<gr(Bh*ADA6), 256, 0, stream>>>(stb, ada_w, ada_b, adab);
  k_ln_mod<<<NB, 128, 0, stream>>>(s, adab, 0, CS, s_mod, 1e-6f);

  // ---- IPA projections (bf16 weights, async-LDS B path) ----
  gemm_t<__bf16>(stream, s_mod,0,0,CS, wq,0,0,CS,1,   q_b,  qb,0,0,192,  NB,192,CS, 1,1, 1.f,0);
  gemm_t<__bf16>(stream, s_mod,0,0,CS, wkv,0,0,CS,1,  kv_b, kvb,0,0,384, NB,384,CS, 1,1, 1.f,0);
  gemm_t<__bf16>(stream, s_mod,0,0,CS, wqp,0,0,CS,1,  qp_b, qpb,0,0,144, NB,144,CS, 1,1, 1.f,0);
  gemm_t<__bf16>(stream, s_mod,0,0,CS, wkvp,0,0,CS,1, kvp_b,kvpb,0,0,432,NB,432,CS, 1,1, 1.f,0);
  gemm_t<__bf16>(stream, z,0,0,CZ,     wbz,0,0,CZ,1,  b_b,  bzb,0,0,HI,  Bh*Nn*Nn,HI,CZ, 1,1, 1.f,0);

  k_to_global<<<gr((long long)NB*48), 256, 0, stream>>>(qpb, rot, trans, qpts, 48);
  k_to_global<<<gr((long long)NB*144),256, 0, stream>>>(kvpb,rot, trans, kvpts,144);
  k_feat<<<gr(Bh*HI*Nn), 256, 0, stream>>>(qb, kvb, qpts, kvpts, head_w, afeat, bfeat, knb);

  // ---- fused IPA logits: one K=32 WMMA per tile (q.k + point cross-term) ----
  gemm_t<float>(stream, afeat,(long long)Nn*32,0,32,
                bfeat,(long long)Nn*32,0,32,1, nullptr,
                scores,(long long)Nn*Nn,0,Nn, Nn,Nn,32, Bh*HI,1, 1.f,0);
  k_softmax_ipa<<<Bh*HI*Nn, 128, 0, stream>>>(scores, knb, bzb, mask);

  // ---- attention-weighted sums ----
  gemm_t<float>(stream, scores,(long long)HI*Nn*Nn,(long long)Nn*Nn,Nn,
                kvb+16,(long long)Nn*384,32,384,0, nullptr,
                oh,(long long)HI*Nn*16,(long long)Nn*16,16, Nn,16,Nn, Bh*HI,HI, 1.f,0);
  gemm_t<float>(stream, scores,(long long)HI*Nn*Nn,(long long)Nn*Nn,Nn,
                kvpts+12,(long long)Nn*432,36,432,0, nullptr,
                optg,(long long)HI*Nn*24,(long long)Nn*24,24, Nn,24,Nn, Bh*HI,HI, 1.f,0);
  gemm_t<float>(stream, scores,(long long)HI*Nn*Nn,Nn,Nn*Nn,
                z,(long long)Nn*Nn*CZ,(long long)Nn*CZ,CZ,0, nullptr,
                opair,(long long)Nn*HI*CZ,(long long)HI*CZ,CZ, HI,CZ,Nn, NB,Nn, 1.f,0);

  k_cat_o<<<gr((long long)NB*192), 256, 0, stream>>>(oh, cat);
  k_cat_pts<<<gr((long long)NB*HI*PV), 256, 0, stream>>>(optg, rot, trans, cat);
  k_cat_opair<<<gr((long long)NB*1536), 256, 0, stream>>>(opair, cat);

  gemm_t<__bf16>(stream, cat,0,0,2112, wout,0,0,2112,1, out_b, ipaout,0,0,CS, NB,CS,2112, 1,1, 1.f,0);
  k_gate_add_ln<<<NB, 128, 0, stream>>>(s, ipaout, adab, mask, ln_ipa_g, ln_ipa_b, spost, scat);

  // ---- s_cat skip features ----
  gemm_t<__bf16>(stream, s_init,0,0,CS, wskip,0,0,CS,1, skip_b, scat+384,0,0,DTR, NB,64,CS, 1,1, 1.f,0);
  gemm_t<__bf16>(stream, dssp,0,0,8,    wsd,0,0,8,1,    sd_b,   scat+448,0,0,DTR, NB,64,8,  1,1, 1.f,0);

  // ---- transformer encoder (2 layers, D=512, 8 heads x 64) ----
  for (int l = 0; l < 2; ++l) {
    gemm_t<__bf16>(stream, scat,0,0,DTR, wqkv + (long long)l*DTR*1536,0,0,DTR,1,
                   tr_qkv_b + l*1536, qkvb,0,0,1536, NB,1536,DTR, 1,1, 1.f,0);
    gemm_t<float>(stream, qkvb,(long long)Nn*1536,64,1536,
                  qkvb+512,(long long)Nn*1536,64,1536,1, nullptr,
                  att,(long long)HTR*Nn*Nn,(long long)Nn*Nn,Nn,
                  Nn,Nn,64, Bh*HTR,HTR, 0.125f,0);
    k_softmax_mask<<<Bh*HTR*Nn, 128, 0, stream>>>(att, mask);
    gemm_t<float>(stream, att,(long long)HTR*Nn*Nn,(long long)Nn*Nn,Nn,
                  qkvb+1024,(long long)Nn*1536,64,1536,0, nullptr,
                  attno,(long long)Nn*DTR,64,DTR, Nn,64,Nn, Bh*HTR,HTR, 1.f,0);
    gemm_t<__bf16>(stream, attno,0,0,DTR, wo + (long long)l*DTR*DTR,0,0,DTR,1,
                   tr_o_b + l*DTR, t1,0,0,DTR, NB,DTR,DTR, 1,1, 1.f,0);
    k_add_ln<<<NB, 128, 0, stream>>>(scat, t1, tr_ln1_g + l*DTR, tr_ln1_b + l*DTR);
    gemm_t<__bf16>(stream, scat,0,0,DTR, wf1 + (long long)l*DTR*DTR,0,0,DTR,1,
                   tr_ff1_b + l*DTR, t1,0,0,DTR, NB,DTR,DTR, 1,1, 1.f,1);
    gemm_t<__bf16>(stream, t1,0,0,DTR, wf2 + (long long)l*DTR*DTR,0,0,DTR,1,
                   tr_ff2_b + l*DTR, t2,0,0,DTR, NB,DTR,DTR, 1,1, 1.f,0);
    k_add_ln<<<NB, 128, 0, stream>>>(scat, t2, tr_ln2_g + l*DTR, tr_ln2_b + l*DTR);
  }

  // ---- post-transformer projection + adaLN + node-transition MLP ----
  gemm_t<__bf16>(stream, scat,0,0,DTR, wpt,0,0,DTR,1, pt_b, ptout,0,0,CS, NB,CS,DTR, 1,1, 1.f,0);
  k_add<<<gr((long long)NB*CS), 256, 0, stream>>>(spost, ptout, s2b, NB*CS);
  k_ln_mod<<<NB, 128, 0, stream>>>(s2b, adab, 3*CS, 4*CS, hb, 1e-6f);
  gemm_t<__bf16>(stream, hb,0,0,CS, wn1,0,0,CS,1, nt1_b, a1,0,0,CS, NB,CS,CS, 1,1, 1.f,1);
  gemm_t<__bf16>(stream, a1,0,0,CS, wn2,0,0,CS,1, nt2_b, a2,0,0,CS, NB,CS,CS, 1,1, 1.f,1);
  gemm_t<__bf16>(stream, a2,0,0,CS, wn3,0,0,CS,1, nt3_b, a3,0,0,CS, NB,CS,CS, 1,1, 1.f,0);
  k_nt_final<<<gr((long long)NB*CS), 256, 0, stream>>>(s2b, hb, a3, adab, a1); // a1 := s3
  k_final_ln<<<NB, 128, 0, stream>>>(a1, ln_tr_g, ln_tr_b, mask, dout);

  // ---- rigid update ----
  gemm_t<__bf16>(stream, dout,0,0,CS, wbb,0,0,CS,1, bb_b, updb,0,0,6, NB,6,CS, 1,1, 1.f,0);
  k_rigid<<<gr(NB), 256, 0, stream>>>(updb, rot, trans, dout + OUT_ROT, dout + OUT_TR);

  // ---- z passthrough (151 MB, L2-resident streaming copy) ----
  k_copy4<<<gr(Z_CNT/4), 256, 0, stream>>>((const float4*)z, (float4*)(dout + OUT_Z), Z_CNT/4);
}
